// RNNLSTM_70171175682679
// MI455X (gfx1250) — matrix operand order repaired and checked by reading, not compiled, not run on hardware
//
#include <hip/hip_runtime.h>
#include <hip/hip_bf16.h>

// ---------------------------------------------------------------------------
// CDNA5 (gfx1250) LSTM: bf16 WMMA GEMMs, TDM + async global->LDS staging,
// register double-buffered k-loops, fused per-timestep recurrence.
// ---------------------------------------------------------------------------

typedef __bf16 bf16_t;
typedef __attribute__((ext_vector_type(16))) __bf16 v16bf;
typedef __attribute__((ext_vector_type(8)))  __bf16 v8bf;
typedef __attribute__((ext_vector_type(8)))  float  v8f;
typedef __attribute__((ext_vector_type(4)))  unsigned int v4u;
typedef __attribute__((ext_vector_type(8)))  int v8i_;
typedef __attribute__((ext_vector_type(4)))  int v4i_;

union FragBF {
    v16bf f;
    v8bf  h[2];
};

#define LSTM_L    2
#define LSTM_BS   32
#define LSTM_NT   256
#define LSTM_D    1024
#define LSTM_H    1024
#define LSTM_G    (4 * LSTM_H)             // 4096 gate columns
#define LSTM_ROWS (LSTM_NT * LSTM_BS)      // 8192 rows (n*BS + b)

__device__ __forceinline__ float sigmoidf_(float x) {
    return 1.0f / (1.0f + __expf(-x));
}

// A fragment (16x32 bf16, ISA 7.12.2): lane holds row l%16; VGPR0-3 = K
// (l/16)*8+0..7, VGPR4-7 = K 16+(l/16)*8+0..7 -> two contiguous 16B loads.
__device__ __forceinline__ FragBF load_frag_a(const bf16_t* row, int k, int half) {
    FragBF a;
    a.h[0] = *reinterpret_cast<const v8bf*>(row + k + half * 8);
    a.h[1] = *reinterpret_cast<const v8bf*>(row + k + 16 + half * 8);
    return a;
}

// B fragment (32x16 bf16): lane holds col l%16; VGPR0-3 = K (l/16)*8+0..7,
// VGPR4-7 = K 16+(l/16)*8+0..7 -> two 16B loads from the weight row.
__device__ __forceinline__ FragBF load_frag_b(const bf16_t* row, int k, int half) {
    FragBF b;
    b.h[0] = *reinterpret_cast<const v8bf*>(row + k + half * 8);
    b.h[1] = *reinterpret_cast<const v8bf*>(row + k + 16 + half * 8);
    return b;
}

// CDNA5 async copy: per-lane global -> LDS, tracked by ASYNCcnt.
// Flat->LDS aperture keeps the LDS byte offset in addr[31:0] (ISA 10.2),
// so truncating the generic pointer yields the LDS address VGPR value.
__device__ __forceinline__ void async_copy_b128_to_lds(void* lds_ptr,
                                                       const void* gptr) {
    uint32_t laddr = (uint32_t)(uintptr_t)lds_ptr;
    uint64_t gaddr = (uint64_t)(uintptr_t)gptr;
    asm volatile("global_load_async_to_lds_b128 %0, %1, off"
                 :: "v"(laddr), "v"(gaddr)
                 : "memory");
}

__device__ __forceinline__ void wait_async_all() {
    asm volatile("s_wait_asynccnt 0x0" ::: "memory");
}

// Tensor Data Mover: 1-D contiguous tile (elems x 2-byte) global -> LDS.
// D# packing per ISA 8.3/8.4: group0 {count=1, lds_addr[63:32],
// global_addr[120:64], type=2[127:126]}; group1 {data_size=1 (2B) [17:16],
// tensor_dim0[79:48], tensor_dim1[111:80]=1, tile_dim0[127:112],
// tensor_dim0_stride[207:160]}. Groups 2/3 zero (tensor <= 2-D).
// This toolchain exposes the 6-arg builtin (g0, g1, g2, g3, g4, cpol).
__device__ __forceinline__ void tdm_load_1d_to_lds(void* lds_ptr,
                                                   const void* gptr,
                                                   unsigned elems) {
    const unsigned lds_addr = (uint32_t)(uintptr_t)lds_ptr;
    const unsigned long long ga = (unsigned long long)(uintptr_t)gptr;
    v4u g0;
    g0[0] = 1u;                                   // count=1 (valid), user mode
    g0[1] = lds_addr;                             // lds_addr (bytes)
    g0[2] = (unsigned)(ga & 0xFFFFFFFFu);         // global_addr[31:0]
    g0[3] = (unsigned)((ga >> 32) & 0x01FFFFFFu)  // global_addr[56:32]
          | (2u << 30);                           // type=2 ("image")
    v8i_ g1;
    g1[0] = (int)(1u << 16);                      // data_size=1 -> 2 bytes
    g1[1] = (int)((elems & 0xFFFFu) << 16);       // tensor_dim0[15:0] @ 63:48
    g1[2] = (int)((elems >> 16) & 0xFFFFu)        // tensor_dim0[31:16] @ 79:64
          | (int)(1u << 16);                      // tensor_dim1=1 @ 111:80
    g1[3] = (int)((elems & 0xFFFFu) << 16);       // tile_dim0 @ 127:112
    g1[4] = 0;                                    // tile_dim1=0, tile_dim2=0
    g1[5] = (int)elems;                           // tensor_dim0_stride[31:0]
    g1[6] = 0;
    g1[7] = 0;
    v4i_ gz4 = {0, 0, 0, 0};
    v8i_ gz8 = {0, 0, 0, 0, 0, 0, 0, 0};
    __builtin_amdgcn_tensor_load_to_lds(g0, g1, gz4, gz4, gz8, 0);
}

// ---------------------------------------------------------------------------
// fp32 -> bf16 conversion (grid-stride)
// ---------------------------------------------------------------------------
__global__ void f32_to_bf16_kernel(const float* __restrict__ in,
                                   bf16_t* __restrict__ out, int n) {
    for (int i = blockIdx.x * blockDim.x + threadIdx.x; i < n;
         i += gridDim.x * blockDim.x)
        out[i] = (bf16_t)in[i];
}

// ---------------------------------------------------------------------------
// State init: h (bf16 ping buffer) and c (f32 workspace)
// ---------------------------------------------------------------------------
__global__ void init_state_kernel(const float* __restrict__ h0,
                                  const float* __restrict__ c0,
                                  bf16_t* __restrict__ h_bf,
                                  float* __restrict__ c_ws, int n) {
    int i = blockIdx.x * blockDim.x + threadIdx.x;
    if (i < n) {
        h_bf[i] = (bf16_t)h0[i];
        c_ws[i] = c0[i];
    }
}

// ---------------------------------------------------------------------------
// GEMM: C[M,N] = A[M,1024] @ B[N,1024]^T + bias1 + bias2.  bf16 in, f32 out.
// Block = 4 waves, all sharing one 16-row A slab. The slab (32 KB,
// contiguous) is staged to LDS by a single TDM descriptor issued by wave 0,
// then each wave computes a 16x64 tile with a register double-buffered
// k-loop so VMEM overlaps WMMA (partial s_wait_loadcnt).
// ---------------------------------------------------------------------------
__global__ __launch_bounds__(128) void gemm_bf16_nt_kernel(
    const bf16_t* __restrict__ A, const bf16_t* __restrict__ B,
    const float* __restrict__ bias1, const float* __restrict__ bias2,
    float* __restrict__ C, int M, int N) {
    constexpr int K = LSTM_D;  // both GEMMs have K = 1024
    __shared__ bf16_t a_lds[16 * K];  // 32 KB

    const int tid  = threadIdx.x;
    const int lane = tid & 31;
    const int wave = tid >> 5;
    const int l16  = lane & 15;
    const int half = lane >> 4;

    const int ntGroups = N >> 6;  // 64 cols per wave; 4 | ntGroups
    const int waveId   = blockIdx.x * 4 + wave;
    const int mt       = waveId / ntGroups;  // same for all 4 waves in block
    const int ng       = waveId % ntGroups;
    const int n0       = ng * 64;

    // Stage A rows [mt*16, mt*16+16) x K into LDS with one TDM transfer.
    if (wave == 0) {
        tdm_load_1d_to_lds(a_lds, A + (size_t)mt * 16 * K, 16u * K);
        __builtin_amdgcn_s_wait_tensorcnt(0);
    }
    __syncthreads();

    const bf16_t* alr = a_lds + (size_t)l16 * K;  // this lane's A row (LDS)
    const bf16_t* Brow[4];
#pragma unroll
    for (int j = 0; j < 4; ++j)
        Brow[j] = B + (size_t)(n0 + j * 16 + l16) * K;

    v8f acc[4] = {};
    FragBF a_cur = load_frag_a(alr, 0, half);
    FragBF b_cur[4];
#pragma unroll
    for (int j = 0; j < 4; ++j) b_cur[j] = load_frag_b(Brow[j], 0, half);

#pragma unroll 2
    for (int k = 0; k < K - 32; k += 32) {
        FragBF a_nxt = load_frag_a(alr, k + 32, half);
        FragBF b_nxt[4];
#pragma unroll
        for (int j = 0; j < 4; ++j) b_nxt[j] = load_frag_b(Brow[j], k + 32, half);
#pragma unroll
        for (int j = 0; j < 4; ++j)
            acc[j] = __builtin_amdgcn_wmma_f32_16x16x32_bf16(
                false, a_cur.f, false, b_cur[j].f, (short)0, acc[j], false, false);
        a_cur = a_nxt;
#pragma unroll
        for (int j = 0; j < 4; ++j) b_cur[j] = b_nxt[j];
    }
#pragma unroll
    for (int j = 0; j < 4; ++j)
        acc[j] = __builtin_amdgcn_wmma_f32_16x16x32_bf16(
            false, a_cur.f, false, b_cur[j].f, (short)0, acc[j], false, false);

#pragma unroll
    for (int j = 0; j < 4; ++j) {
        const int col = n0 + j * 16 + l16;
        float bv = 0.0f;
        if (bias1) bv += bias1[col];
        if (bias2) bv += bias2[col];
#pragma unroll
        for (int v = 0; v < 8; ++v) {
            const int row = mt * 16 + v + 8 * half;
            C[(size_t)row * N + col] = acc[j][v] + bv;
        }
    }
}

// ---------------------------------------------------------------------------
// One LSTM timestep (one layer). 16 blocks x 4 waves. The block async-stages
// the full hidden state h (32x1024 bf16 = 64 KB) into LDS once and all 4
// waves reuse it (4x less global h traffic on the latency-critical scan).
// Each wave owns 16 hidden columns: gates = xg_t + h_in @ Whh^T for the four
// gate groups (cols g*H + n0), then fused sigmoid/tanh c/h update.
// h ping-pong (h_in/h_out) avoids intra-step RAW across blocks.
// ---------------------------------------------------------------------------
__global__ __launch_bounds__(128) void lstm_step_kernel(
    const bf16_t* __restrict__ h_in,   // 32 x H bf16
    bf16_t* __restrict__ h_out,        // 32 x H bf16
    float* __restrict__ c_ws,          // 32 x H f32
    const bf16_t* __restrict__ Whh,    // 4H x H bf16 (row-major)
    const float* __restrict__ xg_t,    // 32 x 4H f32 (this timestep)
    bf16_t* __restrict__ ys_bf,        // 32 x H bf16 (layer output at t)
    float* __restrict__ hf_out,        // 32 x H f32 final h (if write_final)
    float* __restrict__ cf_out,        // 32 x H f32 final c (if write_final)
    int write_final) {
    __shared__ bf16_t h_lds[LSTM_BS * LSTM_H];  // 64 KB

    const int tid  = threadIdx.x;
    const int lane = tid & 31;
    const int wave = tid >> 5;
    const int l16  = lane & 15;
    const int half = lane >> 4;

    // Stage h into LDS (64 KB: 128 lanes x 16 B x 32 iters), async path.
    {
        const char* src = (const char*)h_in;
        char* dst = (char*)h_lds;
        constexpr int BYTES = LSTM_BS * LSTM_H * (int)sizeof(bf16_t);  // 65536
#pragma unroll
        for (int off = 0; off < BYTES; off += 128 * 16)
            async_copy_b128_to_lds(dst + off + tid * 16, src + off + tid * 16);
        wait_async_all();
        __syncthreads();
    }

    const int n0 = blockIdx.x * 64 + wave * 16;  // 16 hidden cols per wave

    const bf16_t* ar0 = h_lds + (size_t)l16 * LSTM_H;         // rows 0..15
    const bf16_t* ar1 = h_lds + (size_t)(16 + l16) * LSTM_H;  // rows 16..31
    const bf16_t* Brow[4];
#pragma unroll
    for (int g = 0; g < 4; ++g)
        Brow[g] = Whh + (size_t)(g * LSTM_H + n0 + l16) * LSTM_H;

    v8f acc[4][2] = {};
    FragBF a0c = load_frag_a(ar0, 0, half);
    FragBF a1c = load_frag_a(ar1, 0, half);
    FragBF bc[4];
#pragma unroll
    for (int g = 0; g < 4; ++g) bc[g] = load_frag_b(Brow[g], 0, half);

#pragma unroll 2
    for (int k = 0; k < LSTM_H - 32; k += 32) {
        FragBF a0n = load_frag_a(ar0, k + 32, half);
        FragBF a1n = load_frag_a(ar1, k + 32, half);
        FragBF bn[4];
#pragma unroll
        for (int g = 0; g < 4; ++g) bn[g] = load_frag_b(Brow[g], k + 32, half);
#pragma unroll
        for (int g = 0; g < 4; ++g) {
            acc[g][0] = __builtin_amdgcn_wmma_f32_16x16x32_bf16(
                false, a0c.f, false, bc[g].f, (short)0, acc[g][0], false, false);
            acc[g][1] = __builtin_amdgcn_wmma_f32_16x16x32_bf16(
                false, a1c.f, false, bc[g].f, (short)0, acc[g][1], false, false);
        }
        a0c = a0n;
        a1c = a1n;
#pragma unroll
        for (int g = 0; g < 4; ++g) bc[g] = bn[g];
    }
#pragma unroll
    for (int g = 0; g < 4; ++g) {
        acc[g][0] = __builtin_amdgcn_wmma_f32_16x16x32_bf16(
            false, a0c.f, false, bc[g].f, (short)0, acc[g][0], false, false);
        acc[g][1] = __builtin_amdgcn_wmma_f32_16x16x32_bf16(
            false, a1c.f, false, bc[g].f, (short)0, acc[g][1], false, false);
    }

    const int col = n0 + l16;  // hidden unit index j
#pragma unroll
    for (int v = 0; v < 8; ++v) {
#pragma unroll
        for (int mt = 0; mt < 2; ++mt) {
            const int r = mt * 16 + v + 8 * half;  // batch row 0..31
            const size_t gbase = (size_t)r * LSTM_G + col;
            float gi = acc[0][mt][v] + xg_t[gbase + 0 * LSTM_H];
            float gf = acc[1][mt][v] + xg_t[gbase + 1 * LSTM_H];
            float gg = acc[2][mt][v] + xg_t[gbase + 2 * LSTM_H];
            float go = acc[3][mt][v] + xg_t[gbase + 3 * LSTM_H];

            const float iv = sigmoidf_(gi);
            const float fv = sigmoidf_(gf);
            const float gv = tanhf(gg);
            const float ov = sigmoidf_(go);

            const size_t idx = (size_t)r * LSTM_H + col;
            const float cn = fv * c_ws[idx] + iv * gv;
            const float hn = ov * tanhf(cn);
            c_ws[idx] = cn;

            const bf16_t hb = (bf16_t)hn;
            h_out[idx] = hb;
            ys_bf[idx] = hb;
            if (write_final) {
                hf_out[idx] = hn;
                cf_out[idx] = cn;
            }
        }
    }
}

// ---------------------------------------------------------------------------
// Host-side orchestration
// ---------------------------------------------------------------------------
extern "C" void kernel_launch(void* const* d_in, const int* in_sizes, int n_in,
                              void* d_out, int out_size, void* d_ws,
                              size_t ws_size, hipStream_t stream) {
    (void)in_sizes; (void)n_in; (void)out_size; (void)ws_size;

    const float* input_ = (const float*)d_in[0];  // (256,32,1024)
    const float* h0     = (const float*)d_in[1];  // (2,32,1024)
    const float* c0     = (const float*)d_in[2];  // (2,32,1024)
    const float* w_ih   = (const float*)d_in[3];  // (2,4096,1024)
    const float* w_hh   = (const float*)d_in[4];  // (2,4096,1024)
    const float* b_ih   = (const float*)d_in[5];  // (2,4096)
    const float* b_hh   = (const float*)d_in[6];  // (2,4096)
    const float* w_lin  = (const float*)d_in[7];  // (1024,1024)
    const float* b_lin  = (const float*)d_in[8];  // (1024,)
    float* out = (float*)d_out;

    // ---- workspace layout ----
    char* ws = (char*)d_ws;
    const size_t WIH_E  = (size_t)LSTM_L * LSTM_G * LSTM_D;
    const size_t WHH_E  = (size_t)LSTM_L * LSTM_G * LSTM_H;
    const size_t WLIN_E = (size_t)LSTM_D * LSTM_H;
    const size_t X_E    = (size_t)LSTM_ROWS * LSTM_D;
    const size_t XG_E   = (size_t)LSTM_ROWS * LSTM_G;
    const size_t ST_E   = (size_t)LSTM_BS * LSTM_H;

    bf16_t* w_ih_bf  = (bf16_t*)ws;  ws += WIH_E * sizeof(bf16_t);
    bf16_t* w_hh_bf  = (bf16_t*)ws;  ws += WHH_E * sizeof(bf16_t);
    bf16_t* w_lin_bf = (bf16_t*)ws;  ws += WLIN_E * sizeof(bf16_t);
    bf16_t* x0_bf    = (bf16_t*)ws;  ws += X_E * sizeof(bf16_t);
    bf16_t* ys0_bf   = (bf16_t*)ws;  ws += X_E * sizeof(bf16_t);
    bf16_t* ys1_bf   = (bf16_t*)ws;  ws += X_E * sizeof(bf16_t);
    float*  xg       = (float*)ws;   ws += XG_E * sizeof(float);
    bf16_t* hA       = (bf16_t*)ws;  ws += ST_E * sizeof(bf16_t);
    bf16_t* hB       = (bf16_t*)ws;  ws += ST_E * sizeof(bf16_t);
    float*  c_ws     = (float*)ws;   ws += ST_E * sizeof(float);

    // ---- convert weights / input to bf16 ----
    f32_to_bf16_kernel<<<4096, 256, 0, stream>>>(w_ih, w_ih_bf, (int)WIH_E);
    f32_to_bf16_kernel<<<4096, 256, 0, stream>>>(w_hh, w_hh_bf, (int)WHH_E);
    f32_to_bf16_kernel<<<1024, 256, 0, stream>>>(w_lin, w_lin_bf, (int)WLIN_E);
    f32_to_bf16_kernel<<<4096, 256, 0, stream>>>(input_, x0_bf, (int)X_E);

    float* hs_out = out + (size_t)LSTM_ROWS * LSTM_D;  // (2,32,1024)
    float* cs_out = hs_out + (size_t)LSTM_L * ST_E;    // (2,32,1024)

    bf16_t* layer_in[LSTM_L]  = {x0_bf, ys0_bf};
    bf16_t* layer_out[LSTM_L] = {ys0_bf, ys1_bf};

    for (int l = 0; l < LSTM_L; ++l) {
        // xg = x @ w_ih[l]^T + (b_ih[l] + b_hh[l]) : (8192 x 4096)
        {
            const int M = LSTM_ROWS, N = LSTM_G;
            const int blocks = (M / 16) * (N / 64) / 4;
            gemm_bf16_nt_kernel<<<blocks, 128, 0, stream>>>(
                layer_in[l], w_ih_bf + (size_t)l * LSTM_G * LSTM_D,
                b_ih + (size_t)l * LSTM_G, b_hh + (size_t)l * LSTM_G,
                xg, M, N);
        }

        // init h (bf16) and c (f32) from provided initial states
        init_state_kernel<<<(int)(ST_E / 256), 256, 0, stream>>>(
            h0 + (size_t)l * ST_E, c0 + (size_t)l * ST_E, hA, c_ws, (int)ST_E);

        const bf16_t* Whh_l = w_hh_bf + (size_t)l * LSTM_G * LSTM_H;
        for (int t = 0; t < LSTM_NT; ++t) {
            bf16_t* h_in  = (t & 1) ? hB : hA;
            bf16_t* h_out = (t & 1) ? hA : hB;
            lstm_step_kernel<<<LSTM_H / 64, 128, 0, stream>>>(
                h_in, h_out, c_ws, Whh_l,
                xg + (size_t)t * LSTM_BS * LSTM_G,
                layer_out[l] + (size_t)t * LSTM_BS * LSTM_H,
                hs_out + (size_t)l * ST_E, cs_out + (size_t)l * ST_E,
                (t == LSTM_NT - 1) ? 1 : 0);
        }
    }

    // out = ys1 @ w_lin^T + b_lin : (8192 x 1024)
    {
        const int M = LSTM_ROWS, N = LSTM_D;
        const int blocks = (M / 16) * (N / 64) / 4;
        gemm_bf16_nt_kernel<<<blocks, 128, 0, stream>>>(
            ys1_bf, w_lin_bf, b_lin, (const float*)nullptr, out, M, N);
    }
}
